// Encoder_16716012716128
// MI455X (gfx1250) — matrix operand order repaired and checked by reading, not compiled, not run on hardware
//
#include <hip/hip_runtime.h>
#include <math.h>

// ---------------- CDNA5 wave32 WMMA types ----------------
typedef __bf16 bf16_t;
typedef __attribute__((ext_vector_type(16))) bf16_t v16bf;
typedef __attribute__((ext_vector_type(8)))  bf16_t v8bf;
typedef __attribute__((ext_vector_type(8)))  float  v8f;

#define BB 64
#define TT 512
#define HH 512
#define DD 128

// packed-weight element offsets (bf16 elems) inside one direction's block
#define WI_ELEMS   (4 * 96 * 512)                  // Wi: K=128 (4 ktiles) x N=1536 (96 ntiles)
#define WHRZ_BASE  (WI_ELEMS)                      // Whrz: K=512 (16) x N=1024 (64)
#define WHN_BASE   (WI_ELEMS + 16 * 64 * 512)      // Whn: K=512 (16) x N=512 (32)
#define DIR_ELEMS  (WHN_BASE + 16 * 32 * 512)      // = 983040 elems = 1.875 MB per direction

// workspace byte offsets
#define WS_FEAT_OFF  (2ull * DIR_ELEMS * 2ull)     // 3,932,160
#define WS_H1_OFF    (WS_FEAT_OFF + 64ull * 1024ull * 4ull)
#define WS_H2_OFF    (WS_H1_OFF   + 64ull * 512ull  * 4ull)

__device__ __forceinline__ unsigned short f2bf(float f) {
  unsigned int uu = __float_as_uint(f);
  unsigned int r = (uu + 0x7FFFu + ((uu >> 16) & 1u)) >> 16;
  return (unsigned short)r;
}
__device__ __forceinline__ float bf2f(unsigned short b) {
  return __uint_as_float(((unsigned int)b) << 16);
}
__device__ __forceinline__ float sigm(float x) { return 1.0f / (1.0f + __expf(-x)); }

__device__ __forceinline__ v8f wmma_bf16(v16bf a, v16bf b, v8f c) {
  // (neg_a, A, neg_b, B, c_mod, C, reuse_a, reuse_b)
  return __builtin_amdgcn_wmma_f32_16x16x32_bf16(false, a, false, b, (short)0, c, false, false);
}

// B fragment: lane's 16 bf16 are contiguous in the packed buffer (32 B per lane)
__device__ __forceinline__ v16bf loadB(const unsigned short* base_elems, int kt, int lane) {
  return ((const v16bf*)(base_elems + kt * 512))[lane];
}

// A fragment from an LDS bf16 row: elements 0..7 -> K=koff+half*8+0..7,
// elements 8..15 -> K=koff+16+half*8+0..7   (ISA 16-bit A 16x32 layout)
__device__ __forceinline__ v16bf loadA(const unsigned short* rowbase, int koff, int half) {
  union { v16bf v; v8bf h2[2]; } a;
  a.h2[0] = *(const v8bf*)(rowbase + koff + half * 8);
  a.h2[1] = *(const v8bf*)(rowbase + koff + 16 + half * 8);
  return a.v;
}

// A fragment from an LDS fp32 row (async-staged x tile), bf16 round on the fly
__device__ __forceinline__ v16bf loadA_f32(const float* rowbase, int koff, int half) {
  union { v8f v; float s[8]; } lo, hi;
  lo.v = *(const v8f*)(rowbase + koff + half * 8);
  hi.v = *(const v8f*)(rowbase + koff + 16 + half * 8);
  union { v16bf v; unsigned short s[16]; } a;
#pragma unroll
  for (int i = 0; i < 8; ++i) {
    a.s[i]     = f2bf(lo.s[i]);
    a.s[8 + i] = f2bf(hi.s[i]);
  }
  return a.v;
}

// ---------------- weight repack: fp32 row-major -> bf16 WMMA-B fragments ----------------
__global__ void repack_weights(const float* Wi_f, const float* Whrz_f, const float* Whn_f,
                               const float* Wi_b, const float* Whrz_b, const float* Whn_b,
                               unsigned short* wpack) {
  int gw   = (blockIdx.x * blockDim.x + threadIdx.x) >> 5;
  int lane = threadIdx.x & 31;
  if (gw >= 3840) return;
  int dir = gw / 1920;
  int f   = gw % 1920;

  const float* src;
  int N, kt, nt;
  unsigned long long dst;
  if (f < 384) {                      // Wi (K=128, N=1536)
    src = dir ? Wi_b : Wi_f;
    N = 1536; nt = f / 4; kt = f % 4;
    dst = (unsigned long long)dir * DIR_ELEMS + (unsigned long long)f * 512ull;
  } else if (f < 1408) {              // Whrz (K=512, N=1024)
    int f2 = f - 384;
    src = dir ? Whrz_b : Whrz_f;
    N = 1024; nt = f2 / 16; kt = f2 % 16;
    dst = (unsigned long long)dir * DIR_ELEMS + WHRZ_BASE + (unsigned long long)f2 * 512ull;
  } else {                            // Whn (K=512, N=512)
    int f3 = f - 1408;
    src = dir ? Whn_b : Whn_f;
    N = 512; nt = f3 / 16; kt = f3 % 16;
    dst = (unsigned long long)dir * DIR_ELEMS + WHN_BASE + (unsigned long long)f3 * 512ull;
  }
  int half = lane >> 4;
  int ncol = nt * 16 + (lane & 15);
#pragma unroll
  for (int e = 0; e < 16; ++e) {
    int k = kt * 32 + (e < 8 ? half * 8 + e : 16 + half * 8 + (e - 8));
    wpack[dst + (unsigned long long)lane * 16ull + e] = f2bf(src[(size_t)k * N + ncol]);
  }
}

// ---------------- fused BiGRU + time-mean pooling ----------------
// grid = 8 blocks (dir + 2*mtile), block = 1024 threads = 32 waves.
// Wave w owns ONE H-column block c=w (16 cols): 4 accumulators (32 VGPRs) leaves
// room to register-double-buffer B fragments so weight loads stay in flight.
// x_t tiles are staged one step ahead via global_load_async_to_lds_b128 (ASYNCcnt).
__global__ void __launch_bounds__(1024)
bigru_kernel(const float* __restrict__ y, const float* __restrict__ u,
             const unsigned short* __restrict__ wpack,
             const float* __restrict__ bi_f, const float* __restrict__ bhn_f,
             const float* __restrict__ bi_b, const float* __restrict__ bhn_b,
             float* __restrict__ feat) {
  const int dir   = blockIdx.x & 1;
  const int mtile = blockIdx.x >> 1;
  const int tid   = threadIdx.x;
  const int lane  = tid & 31;
  const int w     = tid >> 5;        // wave 0..31 -> column block
  const int half  = lane >> 4;
  const int n16   = lane & 15;

  __shared__ __align__(16) unsigned short h_lds[2][16][520];   // hidden state, bf16, double-buffered
  __shared__ __align__(16) float x_stage[2][16][132];          // async-staged x_t tiles (fp32)

  const float* bi  = dir ? bi_b  : bi_f;
  const float* bhn = dir ? bhn_b : bhn_f;
  const unsigned short* wd = wpack + (size_t)dir * DIR_ELEMS;

  const int c   = w;
  const int col = c * 16 + n16;

  const float bir = bi[col];
  const float biz = bi[512 + col];
  const float bin = bi[1024 + col];
  const float bh  = bhn[col];

  // packed B-fragment bases (element offsets; step 512 elems per kt)
  const unsigned short* Bir = wd + (size_t)(c * 4) * 512;
  const unsigned short* Biz = wd + (size_t)((32 + c) * 4) * 512;
  const unsigned short* Bin = wd + (size_t)((64 + c) * 4) * 512;
  const unsigned short* Brr = wd + WHRZ_BASE + (size_t)(c * 16) * 512;
  const unsigned short* Brz = wd + WHRZ_BASE + (size_t)((32 + c) * 16) * 512;
  const unsigned short* Bhn = wd + WHN_BASE  + (size_t)(c * 16) * 512;

  // async stage of one x_t tile [16 x 128] fp32: threads 0..511 copy 16 B each
  auto stage_x = [&](int buf, int tstep) {
    if (tid < 512) {
      int m  = tid >> 5;
      int cc = (tid & 31) * 4;
      int b  = mtile * 16 + m;
      const float* src = (cc < 64) ? (y + ((size_t)b * TT + tstep) * 64 + cc)
                                   : (u + ((size_t)b * TT + tstep) * 64 + (cc - 64));
      unsigned ldsoff = (unsigned)(unsigned long long)(const void*)&x_stage[buf][m][cc];
      asm volatile("global_load_async_to_lds_b128 %0, %1, off"
                   :: "v"(ldsoff), "v"(src) : "memory");
    }
  };

  // h0 = 0; prefetch x(t=0) tile
  for (int i = tid; i < 16 * 520; i += 1024) h_lds[0][i / 520][i % 520] = 0;
  stage_x(0, dir ? TT - 1 : 0);
  v8f hs = {};
  asm volatile("s_wait_asynccnt 0" ::: "memory");
  __syncthreads();

  int cur = 0;
  for (int t = 0; t < TT; ++t) {
    // kick off async stage of next step's x tile while we compute this one
    if (t + 1 < TT) stage_x(cur ^ 1, dir ? (TT - 2 - t) : (t + 1));

    v8f ar = {}, az = {}, ahn = {}, an = {};
    const unsigned short* hrow = &h_lds[cur][n16][0];
    const float* xrow = &x_stage[cur][n16][0];

    // recurrent part: K = 512 over h(t-1); B frags register-double-buffered
    v16bf bR = loadB(Brr, 0, lane);
    v16bf bZ = loadB(Brz, 0, lane);
    v16bf bN = loadB(Bhn, 0, lane);
#pragma unroll
    for (int kt = 0; kt < 16; ++kt) {
      v16bf a = loadA(hrow, kt * 32, half);
      v16bf nR = bR, nZ = bZ, nN = bN;
      if (kt < 15) {
        nR = loadB(Brr, kt + 1, lane);
        nZ = loadB(Brz, kt + 1, lane);
        nN = loadB(Bhn, kt + 1, lane);
      }
      ar  = wmma_bf16(a, bR, ar);
      az  = wmma_bf16(a, bZ, az);
      ahn = wmma_bf16(a, bN, ahn);
      bR = nR; bZ = nZ; bN = nN;
    }
    // input part: K = 128 over x_t (fp32 staged, bf16 round on the fly)
    v16bf iR = loadB(Bir, 0, lane);
    v16bf iZ = loadB(Biz, 0, lane);
    v16bf iN = loadB(Bin, 0, lane);
#pragma unroll
    for (int kt = 0; kt < 4; ++kt) {
      v16bf a = loadA_f32(xrow, kt * 32, half);
      v16bf nR = iR, nZ = iZ, nN = iN;
      if (kt < 3) {
        nR = loadB(Bir, kt + 1, lane);
        nZ = loadB(Biz, kt + 1, lane);
        nN = loadB(Bin, kt + 1, lane);
      }
      ar = wmma_bf16(a, iR, ar);
      az = wmma_bf16(a, iZ, az);
      an = wmma_bf16(a, iN, an);
      iR = nR; iZ = nZ; iN = nN;
    }

    // gates + update (C/D frag: element ri -> row ri + 8*half, col = lane&15)
#pragma unroll
    for (int ri = 0; ri < 8; ++ri) {
      int mr = ri + 8 * half;
      float hp = bf2f(h_lds[cur][mr][col]);
      float r  = sigm(ar[ri] + bir);
      float zz = sigm(az[ri] + biz);
      float nn = tanhf(an[ri] + bin + r * (ahn[ri] + bh));
      float hv = (1.0f - zz) * nn + zz * hp;
      h_lds[cur ^ 1][mr][col] = f2bf(hv);
      hs[ri] += hv;
    }

    // drain this wave's async stage, then one barrier per step
    asm volatile("s_wait_asynccnt 0" ::: "memory");
    __syncthreads();
    cur ^= 1;
  }

  // feat[b, dir*512 + col] = mean_t h
  const float inv = 1.0f / (float)TT;
#pragma unroll
  for (int ri = 0; ri < 8; ++ri) {
    int brow = mtile * 16 + ri + 8 * half;
    feat[(size_t)brow * 1024 + dir * 512 + col] = hs[ri] * inv;
  }
}

// ---------------- tiny MLP head (fp32, negligible FLOPs) ----------------
__global__ void mlp_dense(const float* __restrict__ x, const float* __restrict__ W,
                          const float* __restrict__ b, float* __restrict__ out,
                          int M, int K, int N, int relu) {
  int id = blockIdx.x * blockDim.x + threadIdx.x;
  if (id >= M * N) return;
  int i = id / N, j = id % N;
  float acc = b[j];
  for (int k = 0; k < K; ++k) acc += x[(size_t)i * K + k] * W[(size_t)k * N + j];
  out[id] = relu ? fmaxf(acc, 0.0f) : acc;
}

extern "C" void kernel_launch(void* const* d_in, const int* in_sizes, int n_in,
                              void* d_out, int out_size, void* d_ws, size_t ws_size,
                              hipStream_t stream) {
  const float* y      = (const float*)d_in[0];
  const float* u      = (const float*)d_in[1];
  const float* Wi_f   = (const float*)d_in[2];
  const float* bi_f   = (const float*)d_in[3];
  const float* Whrz_f = (const float*)d_in[4];
  const float* Whn_f  = (const float*)d_in[5];
  const float* bhn_f  = (const float*)d_in[6];
  const float* Wi_b   = (const float*)d_in[7];
  const float* bi_b   = (const float*)d_in[8];
  const float* Whrz_b = (const float*)d_in[9];
  const float* Whn_b  = (const float*)d_in[10];
  const float* bhn_b  = (const float*)d_in[11];
  const float* W1 = (const float*)d_in[12]; const float* b1 = (const float*)d_in[13];
  const float* W2 = (const float*)d_in[14]; const float* b2 = (const float*)d_in[15];
  const float* W3 = (const float*)d_in[16]; const float* b3 = (const float*)d_in[17];

  unsigned short* wpack = (unsigned short*)d_ws;
  float* feat = (float*)((char*)d_ws + WS_FEAT_OFF);
  float* h1   = (float*)((char*)d_ws + WS_H1_OFF);
  float* h2   = (float*)((char*)d_ws + WS_H2_OFF);
  float* out  = (float*)d_out;

  // 1) one-time weight repack into WMMA-B bf16 fragments (3840 wave-fragments)
  repack_weights<<<480, 256, 0, stream>>>(Wi_f, Whrz_f, Whn_f, Wi_b, Whrz_b, Whn_b, wpack);

  // 2) fused bidirectional GRU scan + time-mean (8 WGs: 2 dirs x 4 batch tiles)
  bigru_kernel<<<8, 1024, 0, stream>>>(y, u, wpack, bi_f, bhn_f, bi_b, bhn_b, feat);

  // 3) MLP head
  mlp_dense<<<(64 * 512 + 255) / 256, 256, 0, stream>>>(feat, W1, b1, h1, 64, 1024, 512, 1);
  mlp_dense<<<(64 * 512 + 255) / 256, 256, 0, stream>>>(h1,   W2, b2, h2, 64, 512,  512, 1);
  mlp_dense<<<(64 * 32  + 255) / 256, 256, 0, stream>>>(h2,   W3, b3, out, 64, 512, 32, 0);
}
